// SimpleARRNN_2147483648621
// MI455X (gfx1250) — matrix operand order repaired and checked by reading, not compiled, not run on hardware
//
#include <hip/hip_runtime.h>

// ---------------- problem constants ----------------
#define B_TOT   512
#define LC      512
#define IDIM    64        // input / output feature dim (I == O == 64)
#define HDIM    100       // hidden
#define HK      128       // hidden padded to K multiple of 32
#define NGATE   300       // 3*H
#define NPAD    384       // 3 gate regions padded to 128 each (24 N-tiles of 16)
#define TSTEPS  256
#define TOUT    (TSTEPS + 1)
#define ODIM    64

// ---------------- kernel geometry ----------------
#define BTILE   32                 // batch rows per workgroup
#define NWAVES  (NPAD / 16)        // 24 waves, one N-tile each
#define NTHREADS (NWAVES * 32)     // 768

// LDS strides (floats), odd to avoid bank conflicts on column access
#define GSTR 257
#define NSTR 129
#define ZSTR 65

// Async global->LDS double-buffered context staging (gfx1250 ASYNCcnt path).
#if defined(__HIP_DEVICE_COMPILE__) && \
    __has_builtin(__builtin_amdgcn_global_load_async_to_lds_b128) && \
    __has_builtin(__builtin_amdgcn_s_wait_asynccnt)
#define USE_ASYNC_LDS 1
#else
#define USE_ASYNC_LDS 0
#endif

typedef __attribute__((ext_vector_type(16))) __bf16 v16bf;
typedef __attribute__((ext_vector_type(8)))  __bf16 v8bf;
typedef __attribute__((ext_vector_type(4)))  __bf16 v4bf;
typedef __attribute__((ext_vector_type(8)))  float  v8f;

union V16 { v16bf v; v8bf h[2]; };

#if USE_ASYNC_LDS
// Builtin expects pointers to GCC-vector int4 (per hipcc diagnostic):
//   param0: __device__ (AS1) int4*   param1: __shared__ (AS3) int4*
typedef int asyncv __attribute__((vector_size(4 * sizeof(int))));
typedef __attribute__((address_space(1))) asyncv glob_async_t;
typedef __attribute__((address_space(3))) asyncv lds_async_t;

__device__ __forceinline__ lds_async_t* to_lds(const void* p) {
    // flat LDS addresses truncate to the LDS byte offset (ISA §10.2 aperture)
    return (lds_async_t*)(unsigned int)(unsigned long long)p;
}
__device__ __forceinline__ glob_async_t* to_glob(const void* p) {
    return (glob_async_t*)(unsigned long long)p;
}
#endif

__device__ __forceinline__ v8f wmma_bf16(v16bf a, v16bf b, v8f c) {
    // D = A(16x32 bf16) x B(32x16 bf16) + C(16x16 f32)
    return __builtin_amdgcn_wmma_f32_16x16x32_bf16(
        /*neg_a=*/false, a, /*neg_b=*/false, b,
        /*c_mod=*/(short)0, c, /*reuse_a=*/false, /*reuse_b=*/false);
}

// Load A operand tile (16x32) for k-block kb from row-major bf16 LDS buffer.
// A layout: lanes 0-15 -> M=lane, K = kb*32 + {0..15};
//           lanes 16-31 -> M=lane-16, K = kb*32 + {16..31}.
__device__ __forceinline__ v16bf loadA(const __bf16* base, int stride,
                                       int mrow0, int lane, int kb) {
    const int lm = lane & 15;
    const int g  = lane >> 4;          // 0 or 1
    const __bf16* rp = base + (mrow0 + lm) * stride + kb * 32 + g * 16;
    V16 u;
    u.h[0] = *(const v8bf*)(rp);       // 16B contiguous
    u.h[1] = *(const v8bf*)(rp + 8);   // 16B contiguous
    return u.v;
}

// One GRU pre-activation step: gates = x@Wcomb.T (+chained h@Whh.T for r/z),
// n-region keeps x-part and h-part separate.
__device__ __forceinline__ void do_gru_gemm(
    const __bf16* __restrict__ xs,   // [BTILE][IDIM] bf16 (LDS)
    const __bf16* __restrict__ hb,   // [BTILE][HK]  bf16 (LDS)
    float* __restrict__ grz, float* __restrict__ gxn, float* __restrict__ ghn,
    int w, int lane, const V16* Bcmb, const V16* Bhh)
{
    const int lm = lane & 15;
    const int ro = (lane >= 16) ? 8 : 0;
    for (int m = 0; m < 2; m++) {
        v8f acc = {};
        acc = wmma_bf16(loadA(xs, IDIM, m * 16, lane, 0), Bcmb[0].v, acc);
        acc = wmma_bf16(loadA(xs, IDIM, m * 16, lane, 1), Bcmb[1].v, acc);
        if (w < 16) {
            // r/z regions: chain h-GEMM into same accumulator
            for (int kb = 0; kb < 4; kb++)
                acc = wmma_bf16(loadA(hb, HK, m * 16, lane, kb), Bhh[kb].v, acc);
            const int col = w * 16 + lm;                 // 0..255 (padded r/z)
            for (int r = 0; r < 8; r++)
                grz[(m * 16 + r + ro) * GSTR + col] = acc[r];
        } else {
            // n region: x-part and h-part kept separate (r multiplies nh only)
            const int col = (w - 16) * 16 + lm;          // 0..127 (padded n)
            for (int r = 0; r < 8; r++)
                gxn[(m * 16 + r + ro) * NSTR + col] = acc[r];
            v8f acch = {};
            for (int kb = 0; kb < 4; kb++)
                acch = wmma_bf16(loadA(hb, HK, m * 16, lane, kb), Bhh[kb].v, acch);
            for (int r = 0; r < 8; r++)
                ghn[(m * 16 + r + ro) * NSTR + col] = acch[r];
        }
    }
}

__global__ __launch_bounds__(NTHREADS)
void gru_seq2seq_fused(const float* __restrict__ ctx,   // [B][LC][IDIM]
                       const int*   __restrict__ lens,  // [B]
                       const float* __restrict__ Wemb,  // [H][I]
                       const float* __restrict__ bemb,  // [H]
                       const float* __restrict__ Wih,   // [3H][H]
                       const float* __restrict__ bih,   // [3H]
                       const float* __restrict__ Whh,   // [3H][H]
                       const float* __restrict__ bhh,   // [3H]
                       const float* __restrict__ Wpj,   // [O][H]
                       const float* __restrict__ bpj,   // [O]
                       float* __restrict__ out)         // [B][T+1][O]
{
    __shared__ __attribute__((aligned(16))) __bf16 sXB[BTILE * IDIM];
    __shared__ __attribute__((aligned(16))) __bf16 sZB[BTILE * IDIM];
    __shared__ __attribute__((aligned(16))) __bf16 sHB[BTILE * HK];
#if USE_ASYNC_LDS
    __shared__ __attribute__((aligned(16))) float sXF[2][BTILE * IDIM]; // f32 stage ring
#endif
    __shared__ float sHF[BTILE * HDIM];
    __shared__ float sH0[BTILE * HDIM];
    __shared__ float sGRZ[BTILE * GSTR];
    __shared__ float sGXN[BTILE * NSTR];
    __shared__ float sGHN[BTILE * NSTR];
    __shared__ float sZAC[BTILE * ZSTR];
    __shared__ float sBRZ[200];
    __shared__ float sBXN[HDIM];
    __shared__ float sBHN[HDIM];
    __shared__ float sBPJ[ODIM];
    __shared__ int   sLEN[BTILE];

    const int tid  = threadIdx.x;
    const int lane = tid & 31;
    const int w    = tid >> 5;
    const int lm   = lane & 15;
    const int kh   = (lane >> 4) * 16;       // B-operand K-half this lane owns
    const int b0   = blockIdx.x * BTILE;

    // Gate-reordered padded column owned by this lane's N-slot.
    const int c       = w * 16 + lm;         // 0..383
    const int regn    = c >> 7;              // gate region 0=r,1=z,2=n
    const int within  = c & 127;
    const bool colOK  = (within < HDIM);
    const int nr      = regn * HDIM + within; // original row in [0,300)

    // ---------------- startup: build weight registers ----------------
    V16 Bhh[4], Bcmb[2], Bpj[4];
    for (int kb = 0; kb < 4; kb++)
        for (int e = 0; e < 16; e++) {
            const int k = kb * 32 + kh + e;
            const float v = (colOK && k < HDIM) ? Whh[nr * HDIM + k] : 0.f;
            Bhh[kb].v[e] = (__bf16)v;
        }
    {   // W_comb = W_ih @ W_embed  (row nr), 32 entries per lane
        float accv[32];
        for (int i = 0; i < 32; i++) accv[i] = 0.f;
        if (colOK) {
            for (int hh = 0; hh < HDIM; hh++) {
                const float wv = Wih[nr * HDIM + hh];
                const float* wr = &Wemb[hh * IDIM];
                for (int i = 0; i < 32; i++)
                    accv[i] += wv * wr[(i >> 4) * 32 + kh + (i & 15)];
            }
        }
        for (int i = 0; i < 32; i++) Bcmb[i >> 4].v[i & 15] = (__bf16)accv[i];
    }
    if (w < 8) {   // projection B-operand: waves 0..7, wave = (nt<<1)|m
        const int n = (w >> 1) * 16 + lm;    // 0..63
        for (int kb = 0; kb < 4; kb++)
            for (int e = 0; e < 16; e++) {
                const int k = kb * 32 + kh + e;
                Bpj[kb].v[e] = (__bf16)((k < HDIM) ? Wpj[n * HDIM + k] : 0.f);
            }
    } else {
        for (int kb = 0; kb < 4; kb++)
            for (int e = 0; e < 16; e++) Bpj[kb].v[e] = (__bf16)0.f;
    }

    // ---------------- startup: biases / lengths / zero state ----------------
    if (tid < NGATE) {
        float s = bih[tid];
        for (int hh = 0; hh < HDIM; hh++) s += Wih[tid * HDIM + hh] * bemb[hh];
        if (tid < 200) sBRZ[tid] = s + bhh[tid];   // r/z: x+h biases combined
        else           sBXN[tid - 200] = s;        // n: x-part bias
    } else if (tid < 400) {
        sBHN[tid - 300] = bhh[200 + (tid - 300)];  // n: h-part bias
    } else if (tid < 400 + ODIM) {
        sBPJ[tid - 400] = bpj[tid - 400];
    } else if (tid < 400 + ODIM + BTILE) {
        sLEN[tid - 400 - ODIM] = lens[b0 + (tid - 400 - ODIM)];
    }
    for (int i = tid; i < BTILE * HK; i += NTHREADS) sHB[i] = (__bf16)0.f;
    for (int i = tid; i < BTILE * HDIM; i += NTHREADS) { sHF[i] = 0.f; sH0[i] = 0.f; }

    // Thread-private staging slot (waves 0..15): one float4 of context each.
    const int srow = tid >> 4;               // 0..31 (valid when tid < 512)
    const int si4  = (tid & 15) << 2;        // 0,4,..,60

#if USE_ASYNC_LDS
    if (tid < 512) {   // prime the pipeline: async-fetch t=0 into ring slot 0
        const float* gp = ctx + ((size_t)(b0 + srow) * LC + 0) * IDIM + si4;
        __builtin_amdgcn_global_load_async_to_lds_b128(
            to_glob(gp), to_lds(&sXF[0][srow * IDIM + si4]), 0, 0);
    }
#endif
    __syncthreads();

    // ---------------- gate nonlinearity + state update ----------------
    auto gates = [&](bool enc, int t) {
        for (int k = 0; k < 5; k++) {
            const int idx = tid + k * NTHREADS;
            if (idx < BTILE * HDIM) {
                const int row = idx / HDIM, j = idx - row * HDIM;
                const float r = 1.f / (1.f + __expf(-(sGRZ[row * GSTR + j]       + sBRZ[j])));
                const float z = 1.f / (1.f + __expf(-(sGRZ[row * GSTR + 128 + j] + sBRZ[100 + j])));
                const float n = tanhf(sGXN[row * NSTR + j] + sBXN[j] +
                                      r * (sGHN[row * NSTR + j] + sBHN[j]));
                const float hn = (1.f - z) * n + z * sHF[idx];
                sHF[idx] = hn;
                sHB[row * HK + j] = (__bf16)hn;
                if (enc && t == sLEN[row] - 1) sH0[idx] = hn;
            }
        }
    };

    // ================= ENCODE =================
    for (int t = 0; t < LC; t++) {
        if (tid < 512) {   // stage context[:, t, :] -> bf16 LDS
            float4 v;
#if USE_ASYNC_LDS
            // Wait for this wave's async fill of ring[t&1], then kick off t+1
            // into the other ring slot (fully thread-private: no extra barrier).
            __builtin_amdgcn_s_wait_asynccnt(0);
            if (t + 1 < LC) {
                const float* gp = ctx + ((size_t)(b0 + srow) * LC + (t + 1)) * IDIM + si4;
                __builtin_amdgcn_global_load_async_to_lds_b128(
                    to_glob(gp),
                    to_lds(&sXF[(t + 1) & 1][srow * IDIM + si4]), 0, 0);
            }
            v = *(const float4*)&sXF[t & 1][srow * IDIM + si4];
#else
            const float* p = ctx + ((size_t)(b0 + srow) * LC + t) * IDIM + si4;
            v = *(const float4*)p;
            if (t + 8 < LC) __builtin_prefetch(p + 8 * IDIM, 0, 1); // global_prefetch_b8
#endif
            v4bf bx; bx[0] = (__bf16)v.x; bx[1] = (__bf16)v.y;
                     bx[2] = (__bf16)v.z; bx[3] = (__bf16)v.w;
            *(v4bf*)&sXB[srow * IDIM + si4] = bx;
        }
        __syncthreads();
        do_gru_gemm(sXB, sHB, sGRZ, sGXN, sGHN, w, lane, Bcmb, Bhh);
        __syncthreads();
        gates(true, t);
        __syncthreads();
    }

    // h <- h0 (last valid hidden per row)
    for (int i = tid; i < BTILE * HDIM; i += NTHREADS) {
        const float v = sH0[i];
        sHF[i] = v;
        const int row = i / HDIM, j = i - row * HDIM;
        sHB[row * HK + j] = (__bf16)v;
    }
    __syncthreads();

    // ================= DECODE (autoregressive) =================
    for (int tt = 0; tt <= TSTEPS; tt++) {
        if (w < 8) {   // z = h @ Wproj.T   ([32x128]@[128x64])
            const int m = w & 1, nt = w >> 1;
            const int ro = (lane >= 16) ? 8 : 0;
            v8f acc = {};
            for (int kb = 0; kb < 4; kb++)
                acc = wmma_bf16(loadA(sHB, HK, m * 16, lane, kb), Bpj[kb].v, acc);
            const int col = nt * 16 + lm;
            for (int r = 0; r < 8; r++)
                sZAC[(m * 16 + r + ro) * ZSTR + col] = acc[r];
        }
        __syncthreads();
        for (int k = 0; k < 3; k++) {   // finalize z -> out + bf16 feedback
            const int idx = tid + k * NTHREADS;
            if (idx < BTILE * ODIM) {
                const int row = idx >> 6, o = idx & 63;
                const float zv = sZAC[row * ZSTR + o] + sBPJ[o];
                out[((size_t)(b0 + row) * TOUT + tt) * ODIM + o] = zv;
                sZB[row * IDIM + o] = (__bf16)zv;
            }
        }
        __syncthreads();
        if (tt == TSTEPS) break;
        do_gru_gemm(sZB, sHB, sGRZ, sGXN, sGHN, w, lane, Bcmb, Bhh);
        __syncthreads();
        gates(false, 0);
        __syncthreads();
    }
}

extern "C" void kernel_launch(void* const* d_in, const int* in_sizes, int n_in,
                              void* d_out, int out_size, void* d_ws, size_t ws_size,
                              hipStream_t stream) {
    const float* ctx  = (const float*)d_in[0];
    const int*   lens = (const int*)d_in[1];
    // d_in[2] = t_steps (fixed 256 in reference)
    const float* Wemb = (const float*)d_in[3];
    const float* bemb = (const float*)d_in[4];
    const float* Wih  = (const float*)d_in[5];
    const float* bih  = (const float*)d_in[6];
    const float* Whh  = (const float*)d_in[7];
    const float* bhh  = (const float*)d_in[8];
    const float* Wpj  = (const float*)d_in[9];
    const float* bpj  = (const float*)d_in[10];
    float* out = (float*)d_out;

    hipLaunchKernelGGL(gru_seq2seq_fused,
                       dim3(B_TOT / BTILE), dim3(NTHREADS), 0, stream,
                       ctx, lens, Wemb, bemb, Wih, bih, Whh, bhh, Wpj, bpj, out);
}
